// Gauss3D_54133767799286
// MI455X (gfx1250) — matrix operand order repaired and checked by reading, not compiled, not run on hardware
//
#include <hip/hip_runtime.h>
#include <hip/hip_bf16.h>

// ---------------------------------------------------------------------------
// Gauss-Seidel patch solver + overlap-add reconstruct + conv(1->16)-relu-conv(16->1)
// MI455X / gfx1250: wave32; GS streams A with global_load_b128 (512B/wave/inst);
// conv1 runs as implicit GEMM on v_wmma_f32_16x16x32_f16.
// ---------------------------------------------------------------------------

#define DHW   96
#define VOX   (96*96*96)        // 884736
#define PDIM  48
#define PATCH (48*48*48)        // 110592
#define NP    27
#define EPSF  1e-8f

typedef __attribute__((ext_vector_type(16))) _Float16 v16h;
typedef __attribute__((ext_vector_type(8)))  float    v8f;

__device__ __forceinline__ float softplus_f(float x) {
    return fmaxf(x, 0.f) + log1pf(expf(-fabsf(x)));
}

// ---------------------------------------------------------------------------
// Kernel 1: per-position Gauss-Seidel sweep, 4 positions per thread (float4).
// A reads become global_load_b128, perfectly coalesced & 16B aligned
// (PD=48 % 4 == 0, so a float4 never crosses a patch row).
// cur[27] (float4) stays in VGPRs; cur[i] accessed via unrolled select.
// ---------------------------------------------------------------------------
__global__ void gs_sweep_kernel(const float* __restrict__ x,
                                const float* __restrict__ b,
                                const float* __restrict__ A,
                                float* __restrict__ curbuf) {
    const int q  = blockIdx.x * blockDim.x + threadIdx.x;   // 0..PATCH/4-1 (exact)
    const int p  = q * 4;
    const int pw = p % PDIM;                                // multiple of 4
    const int tt = p / PDIM;
    const int ph = tt % PDIM;
    const int pd = tt / PDIM;

    float4 cur[NP];
    // extract(x): cur[n] = x[LIN[n, p..p+3]]  (contiguous in w, 16B aligned)
#pragma unroll
    for (int n = 0; n < NP; ++n) {
        const int od = (n / 9) * 24, oh = ((n / 3) % 3) * 24, ow = (n % 3) * 24;
        cur[n] = *reinterpret_cast<const float4*>(
            x + ((od + pd) * DHW + (oh + ph)) * DHW + (ow + pw));
    }

    for (int i = 0; i < NP; ++i) {
        const float* Ai = A + (size_t)i * NP * PATCH + p;
        // prefetch the next GS row of A (gfx1250 global_prefetch path)
        if (i < NP - 1)
            __builtin_prefetch(Ai + (size_t)NP * PATCH, 0, 3);

        const int od = (i / 9) * 24, oh = ((i / 3) % 3) * 24, ow = (i % 3) * 24;
        const float4 bi = *reinterpret_cast<const float4*>(
            b + ((od + pd) * DHW + (oh + ph)) * DHW + (ow + pw));

        float4 cross = make_float4(0.f, 0.f, 0.f, 0.f);
        float4 Aii   = make_float4(0.f, 0.f, 0.f, 0.f);
        float4 curi  = make_float4(0.f, 0.f, 0.f, 0.f);
#pragma unroll
        for (int n = 0; n < NP; ++n) {
            const float4 a = *reinterpret_cast<const float4*>(Ai + (size_t)n * PATCH);
            cross.x = fmaf(a.x, cur[n].x, cross.x);
            cross.y = fmaf(a.y, cur[n].y, cross.y);
            cross.z = fmaf(a.z, cur[n].z, cross.z);
            cross.w = fmaf(a.w, cur[n].w, cross.w);
            if (n == i) { Aii = a; curi = cur[n]; }     // select, stays in VGPRs
        }
        float4 xn;
        xn.x = (bi.x - (cross.x - Aii.x * curi.x)) / (softplus_f(Aii.x) + EPSF);
        xn.y = (bi.y - (cross.y - Aii.y * curi.y)) / (softplus_f(Aii.y) + EPSF);
        xn.z = (bi.z - (cross.z - Aii.z * curi.z)) / (softplus_f(Aii.z) + EPSF);
        xn.w = (bi.w - (cross.w - Aii.w * curi.w)) / (softplus_f(Aii.w) + EPSF);
#pragma unroll
        for (int n = 0; n < NP; ++n)
            if (n == i) cur[n] = xn;                    // select-write, no scratch
    }

#pragma unroll
    for (int n = 0; n < NP; ++n)
        *reinterpret_cast<float4*>(curbuf + (size_t)n * PATCH + p) = cur[n];
}

// ---------------------------------------------------------------------------
// Kernel 2: overlap-add reconstruct (gather form, no atomics) + divide by count.
// ---------------------------------------------------------------------------
__global__ void recon_kernel(const float* __restrict__ curbuf,
                             float* __restrict__ xr) {
    const int v = blockIdx.x * blockDim.x + threadIdx.x;    // exact grid = VOX
    const int w = v % DHW;
    const int t = v / DHW;
    const int h = t % DHW;
    const int d = t / DHW;

    float sum = 0.f;
    int   cnt = 0;
#pragma unroll
    for (int id = 0; id < 3; ++id) {
        const int pd = d - 24 * id;
        if (pd < 0 || pd >= PDIM) continue;
#pragma unroll
        for (int ih = 0; ih < 3; ++ih) {
            const int ph = h - 24 * ih;
            if (ph < 0 || ph >= PDIM) continue;
#pragma unroll
            for (int iw = 0; iw < 3; ++iw) {
                const int pw = w - 24 * iw;
                if (pw < 0 || pw >= PDIM) continue;
                const int n = (id * 3 + ih) * 3 + iw;
                const int p = (pd * PDIM + ph) * PDIM + pw;
                sum += curbuf[(size_t)n * PATCH + p];
                ++cnt;
            }
        }
    }
    xr[v] = sum / (float)cnt;
}

// ---------------------------------------------------------------------------
// Kernel 3: conv(1->16, 3x3x3, pad 1) + bias + ReLU as implicit GEMM on WMMA.
// One wave = 16 consecutive voxels (M) x 16 channels (N), K = 27 taps -> 32.
// Operand layouts per CDNA5 ISA (wave32). EXEC full at the WMMA.
// ---------------------------------------------------------------------------
__global__ void conv1_wmma_kernel(const float* __restrict__ xr,
                                  const float* __restrict__ W1,
                                  const float* __restrict__ b1,
                                  float* __restrict__ y1) {
    const int lane = threadIdx.x & 31;
    const int wave = threadIdx.x >> 5;
    const int tile = blockIdx.x * (blockDim.x >> 5) + wave; // exact: VOX/16 tiles
    const int nCh  = lane & 15;    // B/C/D column
    const int grp  = lane >> 4;    // lane-half selector

    // B operand: weights W1[n][k]; K layout = e<8 ? grp*8 + e : 16 + grp*8 + (e-8)
    v16h bmat;
#pragma unroll
    for (int e = 0; e < 16; ++e) {
        const int k = (e < 8) ? (grp * 8 + e) : (16 + grp * 8 + (e - 8));
        const float wv = (k < 27) ? W1[nCh * 27 + k] : 0.f;
        bmat[e] = (_Float16)wv;
    }

    // C operand: bias broadcast (D = A*B + C fuses the bias add)
    const float bias = b1[nCh];
    v8f cacc;
#pragma unroll
    for (int r = 0; r < 8; ++r) cacc[r] = bias;

    // A operand: im2col row for voxel m = lane&15 of this tile
    const int m  = lane & 15;
    const int v0 = tile * 16;
    const int vm = v0 + m;
    const int w  = vm % DHW;
    const int t2 = vm / DHW;
    const int h  = t2 % DHW;
    const int d  = t2 / DHW;

    v16h amat;
#pragma unroll
    for (int e = 0; e < 16; ++e) {
        const int k = (e < 8) ? (grp * 8 + e) : (16 + grp * 8 + (e - 8));
        float val = 0.f;
        if (k < 27) {
            const int kd = k / 9, kh = (k / 3) % 3, kw = k % 3;
            const int dd = d + kd - 1, hh = h + kh - 1, ww = w + kw - 1;
            if ((unsigned)dd < (unsigned)DHW && (unsigned)hh < (unsigned)DHW &&
                (unsigned)ww < (unsigned)DHW)
                val = xr[(dd * DHW + hh) * DHW + ww];
        }
        amat[e] = (_Float16)val;
    }

    const v8f dmat = __builtin_amdgcn_wmma_f32_16x16x32_f16(
        /*neg_a=*/false, amat, /*neg_b=*/false, bmat,
        /*c_mod=*/(short)0, cacc, /*reuse_a=*/false, /*reuse_b=*/false);

    // D layout: element r -> (m = r + 8*grp, n = nCh); ReLU, store channel-major
#pragma unroll
    for (int r = 0; r < 8; ++r) {
        const int vr = v0 + r + grp * 8;
        y1[(size_t)nCh * VOX + vr] = fmaxf(dmat[r], 0.f);
    }
}

// ---------------------------------------------------------------------------
// Kernel 4: conv(16->1, 3x3x3, pad 1) + bias. Weights staged in LDS.
// ---------------------------------------------------------------------------
__global__ void conv2_kernel(const float* __restrict__ y1,
                             const float* __restrict__ W2,
                             const float* __restrict__ b2,
                             float* __restrict__ out) {
    __shared__ float w2s[16 * 27];
    for (int idx = threadIdx.x; idx < 16 * 27; idx += blockDim.x)
        w2s[idx] = W2[idx];
    __syncthreads();

    const int v = blockIdx.x * blockDim.x + threadIdx.x;    // exact grid = VOX
    const int w = v % DHW;
    const int t = v / DHW;
    const int h = t % DHW;
    const int d = t / DHW;

    float acc = b2[0];
#pragma unroll 1
    for (int c = 0; c < 16; ++c) {
        const float* __restrict__ yc = y1 + (size_t)c * VOX;
#pragma unroll
        for (int k = 0; k < 27; ++k) {
            const int kd = k / 9, kh = (k / 3) % 3, kw = k % 3;
            const int dd = d + kd - 1, hh = h + kh - 1, ww = w + kw - 1;
            if ((unsigned)dd < (unsigned)DHW && (unsigned)hh < (unsigned)DHW &&
                (unsigned)ww < (unsigned)DHW)
                acc = fmaf(yc[(dd * DHW + hh) * DHW + ww], w2s[c * 27 + k], acc);
        }
    }
    out[v] = acc;
}

// ---------------------------------------------------------------------------
// Launch: ITEM=2 rounds of [GS sweep -> reconstruct -> conv1(WMMA) -> conv2].
// d_out doubles as x_last between iterations (written before it is read).
// ---------------------------------------------------------------------------
extern "C" void kernel_launch(void* const* d_in, const int* in_sizes, int n_in,
                              void* d_out, int out_size, void* d_ws, size_t ws_size,
                              hipStream_t stream) {
    const float* x  = (const float*)d_in[0];
    const float* b  = (const float*)d_in[1];
    const float* A  = (const float*)d_in[2];
    const float* W1 = (const float*)d_in[3];
    const float* b1 = (const float*)d_in[4];
    const float* W2 = (const float*)d_in[5];
    const float* b2 = (const float*)d_in[6];
    float* out = (float*)d_out;

    char* ws = (char*)d_ws;
    float* curbuf = (float*)ws;                                  // 27*PATCH f32 (~12 MB)
    float* xr     = (float*)(ws + (size_t)NP * PATCH * 4);       // VOX f32    (~3.4 MB)
    float* y1     = xr + VOX;                                    // 16*VOX f32 (~54 MB)

    const int gsBlocks    = (PATCH / 4) / 256;  // 108  (exact, 4 pos/thread)
    const int voxBlocks   = VOX / 256;          // 3456 (exact)
    const int conv1Blocks = (VOX / 16) / 8;     // 6912 (8 waves/block, exact)

    for (int it = 0; it < 2; ++it) {
        const float* xsrc = (it == 0) ? x : out;
        gs_sweep_kernel  <<<gsBlocks,    256, 0, stream>>>(xsrc, b, A, curbuf);
        recon_kernel     <<<voxBlocks,   256, 0, stream>>>(curbuf, xr);
        conv1_wmma_kernel<<<conv1Blocks, 256, 0, stream>>>(xr, W1, b1, y1);
        conv2_kernel     <<<voxBlocks,   256, 0, stream>>>(y1, W2, b2, out);
    }
}
